// GatedWaveDynamicsLayer_19207093747870
// MI455X (gfx1250) — matrix operand order repaired and checked by reading, not compiled, not run on hardware
//
#include <hip/hip_runtime.h>
#include <hip/hip_bf16.h>
#include <math.h>

typedef __bf16 bf16;
typedef __attribute__((ext_vector_type(16))) __bf16 v16bf;
typedef __attribute__((ext_vector_type(8)))  float  v8f;

union FragV { v16bf v; uint4 q[2]; };

#define S_LEN  4096
#define DIN    1024
#define DS     256
#define ROWS   16384      // B*S
#define CHUNK  128
#define NCHUNK 32         // S_LEN / CHUNK

// float -> bf16, round-to-nearest-even (storage only; HW consumes raw bits)
__device__ __forceinline__ bf16 f2bf(float f) {
  union { float f; unsigned int u; } v; v.f = f;
  unsigned int r = v.u + 0x7FFFu + ((v.u >> 16) & 1u);
  union { unsigned short s; bf16 b; } o; o.s = (unsigned short)(r >> 16);
  return o.b;
}

// CDNA5 async global->LDS copy (ASYNCcnt-tracked), 16 bytes per lane.
__device__ __forceinline__ void async_load_b128(unsigned ldsoff, const bf16* gptr) {
  asm volatile("global_load_async_to_lds_b128 %0, %1, off"
               :: "v"(ldsoff), "v"(gptr) : "memory");
}
__device__ __forceinline__ void wait_async0() {
  asm volatile("s_wait_asynccnt 0" ::: "memory");
}

__device__ __forceinline__ v8f wmma_bf16(const FragV& a, const FragV& b, v8f c) {
  return __builtin_amdgcn_wmma_f32_16x16x32_bf16(false, a.v, false, b.v,
                                                 (short)0, c, false, false);
}

// ---------------- prep kernels ----------------
__global__ void cvt_x_kernel(const float* __restrict__ x, bf16* __restrict__ xb, int n) {
  int i = blockIdx.x * blockDim.x + threadIdx.x;
  if (i < n) xb[i] = f2bf(x[i]);
}

// Wt[c][d], c = n*768 + sel*256 + k  (column-major weights: K(d) contiguous)
__global__ void pack_w1_kernel(const float* __restrict__ Wg1,
                               const float* __restrict__ Wre,
                               const float* __restrict__ Wim,
                               bf16* __restrict__ Wt) {
  int tid = blockIdx.x * blockDim.x + threadIdx.x;   // 3072*1024
  int d = tid & 1023;
  int c = tid >> 10;
  int n = c / 768; int rr = c - n * 768; int sel = rr >> 8; int k = rr & 255;
  const float* src = (sel == 0) ? Wg1 : (sel == 1) ? Wre : Wim;
  Wt[tid] = f2bf(src[((size_t)n * DIN + d) * DS + k]);
}

// Wt2[n][j][k] = Wg2[n][k][j]
__global__ void pack_w2_kernel(const float* __restrict__ Wg2, bf16* __restrict__ Wt2) {
  int tid = blockIdx.x * blockDim.x + threadIdx.x;   // 4*256*256
  int k = tid & 255; int j = (tid >> 8) & 255; int n = tid >> 16;
  Wt2[tid] = f2bf(Wg2[((size_t)n * DS + k) * DS + j]);
}

// ---------------- shared GEMM core ----------------
// Block: 256 threads = 8 waves. Block tile: 256 rows x 64 cols.
// Each wave: 32 rows (2 M-tiles) x 4 N-tiles -> 8 WMMA per K-step.
// B tile (64 cols x 32 k, 4KB) staged to LDS via async copy, double-buffered.
template <int KDIM>
__device__ __forceinline__ void gemm_block(const bf16* __restrict__ A,
                                           const bf16* __restrict__ Bcols,
                                           int row0, v8f acc[2][4],
                                           bf16 (*bsh)[64][32]) {
  const int tid  = threadIdx.x;
  const int lane = tid & 31;
  const int aKoff = (lane >> 4) << 3;   // lanes>=16: K+8
  const int bKoff = (lane >> 4) << 3;   // lanes>=16: K+8
  const int ncol  = lane & 15;
  const bf16* aP0 = A + (size_t)(row0 + (lane & 15)) * KDIM;
  const bf16* aP1 = aP0 + (size_t)16 * KDIM;

  // staging: thread -> (col = tid/4, 8-half chunk = tid%4); 16B per thread
  const int scol = tid >> 2;
  const int skc  = tid & 3;
  const bf16* gStage = Bcols + (size_t)scol * KDIM + skc * 8;
  const unsigned ldsBase = (unsigned)(uintptr_t)&bsh[0][0][0];
  const unsigned myLds   = ldsBase + (unsigned)(scol * 64 + skc * 16);

  async_load_b128(myLds, gStage);          // stage k0=0 into buf 0
  wait_async0();
  __syncthreads();

  int buf = 0;
  for (int k0 = 0; k0 < KDIM; k0 += 32, buf ^= 1) {
    if (k0 + 32 < KDIM)                    // prefetch next K-step into other buf
      async_load_b128(myLds + (unsigned)((buf ^ 1) * 4096), gStage + k0 + 32);

    // issue all loads first (A from global, all 4 B frags from LDS),
    // then run the 8-WMMA chain without per-pair LDS waits
    FragV a0, a1, b[4];
    a0.q[0] = *reinterpret_cast<const uint4*>(aP0 + k0 + aKoff);
    a0.q[1] = *reinterpret_cast<const uint4*>(aP0 + k0 + 16 + aKoff);
    a1.q[0] = *reinterpret_cast<const uint4*>(aP1 + k0 + aKoff);
    a1.q[1] = *reinterpret_cast<const uint4*>(aP1 + k0 + 16 + aKoff);
#pragma unroll
    for (int t = 0; t < 4; ++t) {
      const bf16* lp = &bsh[buf][t * 16 + ncol][bKoff];
      b[t].q[0] = *reinterpret_cast<const uint4*>(lp);
      b[t].q[1] = *reinterpret_cast<const uint4*>(lp + 16);
    }
#pragma unroll
    for (int t = 0; t < 4; ++t) {
      acc[0][t] = wmma_bf16(a0, b[t], acc[0][t]);
      acc[1][t] = wmma_bf16(a1, b[t], acc[1][t]);
    }
    wait_async0();                          // own prefetch landed
    __syncthreads();                        // whole tile visible / reads done
  }
}

// ---------------- GEMM1: X(16384x1024) * Wt(1024x3072) ----------------
__global__ void __launch_bounds__(256) gemm1_kernel(
    const bf16* __restrict__ Xb, const bf16* __restrict__ Wt,
    const float* __restrict__ bg1, const float* __restrict__ bre,
    const float* __restrict__ bim,
    bf16* __restrict__ g1, float* __restrict__ dre, float* __restrict__ dimg) {
  __shared__ alignas(16) bf16 bsh[2][64][32];
  const int lane = threadIdx.x & 31;
  const int wave = threadIdx.x >> 5;
  const int row0 = blockIdx.x * 256 + wave * 32;
  const int cBlock = blockIdx.y * 64;          // 48 blocks of 64 cols
  const int region = cBlock >> 8;              // n*3+sel
  const int nsc = region / 3;
  const int sel = region - nsc * 3;
  const int kcol0 = cBlock & 255;

  v8f z = {0, 0, 0, 0, 0, 0, 0, 0};
  v8f acc[2][4] = {{z, z, z, z}, {z, z, z, z}};
  gemm_block<DIN>(Xb, Wt + (size_t)cBlock * DIN, row0, acc, bsh);

  const int ncol = lane & 15;
  const int rOff = (lane >> 4) << 3;           // lanes>=16 hold rows M+8
  const float* biasArr = (sel == 0) ? bg1 : (sel == 1) ? bre : bim;
#pragma unroll
  for (int mt = 0; mt < 2; ++mt) {
#pragma unroll
    for (int t = 0; t < 4; ++t) {
      const int col = kcol0 + t * 16 + ncol;
      const float bias = biasArr[nsc * DS + col];
      const size_t base = ((size_t)nsc * ROWS + row0 + mt * 16 + rOff) * DS + col;
#pragma unroll
      for (int r = 0; r < 8; ++r) {
        float v = acc[mt][t][r] + bias;
        size_t idx = base + (size_t)r * DS;
        if (sel == 0)      g1[idx]   = f2bf(fmaxf(v, 0.0f));
        else if (sel == 1) dre[idx]  = v;
        else               dimg[idx] = v;
      }
    }
  }
}

// ---------------- GEMM2: per-scale g1(16384x256) * Wt2(256x256) -> omg ----------------
__global__ void __launch_bounds__(256) gemm2_kernel(
    const bf16* __restrict__ g1, const bf16* __restrict__ Wt2,
    const float* __restrict__ bg2, float* __restrict__ omg) {
  __shared__ alignas(16) bf16 bsh[2][64][32];
  const int lane = threadIdx.x & 31;
  const int wave = threadIdx.x >> 5;
  const int n = blockIdx.z;
  const int row0 = blockIdx.x * 256 + wave * 32;
  const int cBlock = blockIdx.y * 64;

  v8f z = {0, 0, 0, 0, 0, 0, 0, 0};
  v8f acc[2][4] = {{z, z, z, z}, {z, z, z, z}};
  gemm_block<DS>(g1 + (size_t)n * ROWS * DS,
                 Wt2 + ((size_t)n * DS + cBlock) * DS, row0, acc, bsh);

  const int ncol = lane & 15;
  const int rOff = (lane >> 4) << 3;
#pragma unroll
  for (int mt = 0; mt < 2; ++mt) {
#pragma unroll
    for (int t = 0; t < 4; ++t) {
      const int col = cBlock + t * 16 + ncol;
      const float bias = bg2[n * DS + col];
      const size_t base = ((size_t)n * ROWS + row0 + mt * 16 + rOff) * DS + col;
#pragma unroll
      for (int r = 0; r < 8; ++r) {
        float l = acc[mt][t][r] + bias;
        omg[base + (size_t)r * DS] = 1.0f / (1.0f + expf(l));  // 1 - sigmoid(l)
      }
    }
  }
}

// ---------------- scan: h[t] = a[t]*h[t-1] + b[t] (complex), along s ----------------
__device__ __forceinline__ void a_base_n(int n, float& cr, float& ci) {
  const float rv[4] = {1.0f, 0.999f, 0.9495f, 0.9f};
  const float tv[4] = {0.0f, 0.01f, 0.505f, 1.0f};
  cr = rv[n] * cosf(tv[n]);
  ci = rv[n] * sinf(tv[n]);
}

__global__ void scan_chunk_kernel(const float* __restrict__ omg,
                                  const float* __restrict__ dre,
                                  const float* __restrict__ dimg,
                                  float4* __restrict__ comp) {
  int tid = blockIdx.x * blockDim.x + threadIdx.x;   // 16 * 32 * 256
  int k = tid & 255;
  int chunk = (tid >> 8) & 31;
  int bn = tid >> 13;                                // b*4+n
  int b = bn >> 2, n = bn & 3;
  float cr, ci; a_base_n(n, cr, ci);
  size_t base = ((size_t)n * ROWS + b * S_LEN + chunk * CHUNK) * DS + k;
  float Ar = 1.f, Ai = 0.f, Br = 0.f, Bi = 0.f;
  for (int t = 0; t < CHUNK; ++t) {
    size_t idx = base + (size_t)t * DS;
    float g = omg[idx];
    float are = g * cr, aim = g * ci;
    float bre = g * dre[idx] * 0.0625f;
    float bim = g * dimg[idx] * 0.0625f;
    float nAr = Ar * are - Ai * aim;
    float nAi = Ar * aim + Ai * are;
    Ar = nAr; Ai = nAi;
    float nBr = are * Br - aim * Bi + bre;
    float nBi = are * Bi + aim * Br + bim;
    Br = nBr; Bi = nBi;
  }
  comp[(size_t)(bn * NCHUNK + chunk) * DS + k] = make_float4(Ar, Ai, Br, Bi);
}

__global__ void scan_carry_kernel(const float4* __restrict__ comp,
                                  float2* __restrict__ pref) {
  int tid = blockIdx.x * blockDim.x + threadIdx.x;   // 4096 channels
  int k = tid & 255;
  int bn = tid >> 8;
  float hr = 0.f, hi = 0.f;
  for (int c = 0; c < NCHUNK; ++c) {
    size_t i = (size_t)(bn * NCHUNK + c) * DS + k;
    pref[i] = make_float2(hr, hi);
    float4 v = comp[i];
    float nr = v.x * hr - v.y * hi + v.z;
    float ni = v.x * hi + v.y * hr + v.w;
    hr = nr; hi = ni;
  }
}

__global__ void scan_apply_kernel(const float* __restrict__ omg,
                                  const float* __restrict__ dre,
                                  const float* __restrict__ dimg,
                                  const float2* __restrict__ pref,
                                  float* __restrict__ out) {
  int tid = blockIdx.x * blockDim.x + threadIdx.x;
  int k = tid & 255;
  int chunk = (tid >> 8) & 31;
  int bn = tid >> 13;
  int b = bn >> 2, n = bn & 3;
  float cr, ci; a_base_n(n, cr, ci);
  size_t base = ((size_t)n * ROWS + b * S_LEN + chunk * CHUNK) * DS + k;
  float2 h0 = pref[(size_t)(bn * NCHUNK + chunk) * DS + k];
  float hr = h0.x, hi = h0.y;
  size_t obase = ((size_t)b * S_LEN + chunk * CHUNK) * 2048 + n * 512 + k;
  for (int t = 0; t < CHUNK; ++t) {
    size_t idx = base + (size_t)t * DS;
    float g = omg[idx];
    float are = g * cr, aim = g * ci;
    float bre = g * dre[idx] * 0.0625f;
    float bim = g * dimg[idx] * 0.0625f;
    float nr = are * hr - aim * hi + bre;
    float ni = are * hi + aim * hr + bim;
    hr = nr; hi = ni;
    out[obase + (size_t)t * 2048]       = hr;
    out[obase + (size_t)t * 2048 + 256] = hi;
  }
}

extern "C" void kernel_launch(void* const* d_in, const int* in_sizes, int n_in,
                              void* d_out, int out_size, void* d_ws, size_t ws_size,
                              hipStream_t stream) {
  const float* x   = (const float*)d_in[0];
  const float* Wg1 = (const float*)d_in[1];
  const float* bg1 = (const float*)d_in[2];
  const float* Wg2 = (const float*)d_in[3];
  const float* bg2 = (const float*)d_in[4];
  const float* Wre = (const float*)d_in[5];
  const float* bre = (const float*)d_in[6];
  const float* Wim = (const float*)d_in[7];
  const float* bim = (const float*)d_in[8];
  float* out = (float*)d_out;

  char* ws = (char*)d_ws;
  bf16*   xb   = (bf16*)(ws + 0);            // 32 MB
  bf16*   Wt   = (bf16*)(ws + 33554432);     // 6 MB
  bf16*   Wt2  = (bf16*)(ws + 39845888);     // 0.5 MB
  bf16*   g1b  = (bf16*)(ws + 40370176);     // 32 MB
  float*  dreb = (float*)(ws + 73924608);    // 64 MB
  float*  dimb = (float*)(ws + 141033472);   // 64 MB
  float*  omgb = (float*)(ws + 208142336);   // 64 MB
  float4* comp = (float4*)(ws + 275251200);  // 2 MB
  float2* pref = (float2*)(ws + 277348352);  // 1 MB

  cvt_x_kernel<<<16777216 / 256, 256, 0, stream>>>(x, xb, 16777216);
  pack_w1_kernel<<<3145728 / 256, 256, 0, stream>>>(Wg1, Wre, Wim, Wt);
  pack_w2_kernel<<<262144 / 256, 256, 0, stream>>>(Wg2, Wt2);
  gemm1_kernel<<<dim3(64, 48), 256, 0, stream>>>(xb, Wt, bg1, bre, bim,
                                                 g1b, dreb, dimb);
  gemm2_kernel<<<dim3(64, 4, 4), 256, 0, stream>>>(g1b, Wt2, bg2, omgb);
  scan_chunk_kernel<<<512, 256, 0, stream>>>(omgb, dreb, dimb, comp);
  scan_carry_kernel<<<16, 256, 0, stream>>>(comp, pref);
  scan_apply_kernel<<<512, 256, 0, stream>>>(omgb, dreb, dimb, pref, out);
}
